// DynamicSampling_54735063220279
// MI455X (gfx1250) — matrix operand order repaired and checked by reading, not compiled, not run on hardware
//
#include <hip/hip_runtime.h>
#include <math.h>

// Problem constants (match reference)
constexpr int kB = 4;
constexpr int kG = 4;
constexpr int kC = 256;
constexpr int kN = 16384;

constexpr int NT       = 64;    // n-tile per block
constexpr int NTHREADS = 256;   // 8 wave32
constexpr int LDPAD    = 8;     // half-elements of row padding in xT (keeps 16B align)

typedef _Float16 v16h __attribute__((ext_vector_type(16)));
typedef _Float16 v8h  __attribute__((ext_vector_type(8)));
typedef float    v8f  __attribute__((ext_vector_type(8)));

__global__ __launch_bounds__(NTHREADS)
void dynamic_sampling_fused(const float* __restrict__ x,
                            const float* __restrict__ W,
                            const float* __restrict__ bias,
                            const float* __restrict__ u,
                            float* __restrict__ out)
{
    __shared__ _Float16 sWh[16][kC];            // W padded 4->16 rows, f16 (WMMA A)
    __shared__ _Float16 sXT[NT][kC + LDPAD];    // x tile transposed, f16 (WMMA B)
    __shared__ float    sGate[kG][NT];          // hard gate per (g, n_local)

    const int t  = threadIdx.x;
    const int bi = blockIdx.y;
    const int n0 = blockIdx.x * NT;

    // ---- W -> LDS, f16, zero-padded to 16 rows ----
    for (int i = t; i < 16 * kC; i += NTHREADS) {
        const int r = i >> 8;
        const int c = i & (kC - 1);
        sWh[r][c] = (r < kG) ? (_Float16)W[r * kC + c] : (_Float16)0.0f;
    }

    // ---- x tile: f32 -> registers, f16 transposed -> LDS ----
    constexpr int NV4 = (kC * NT) / (4 * NTHREADS);  // 16 float4 per thread
    float4 xv[NV4];
#pragma unroll
    for (int it = 0; it < NV4; ++it) {
        const int i = t + it * NTHREADS;
        const int c = i >> 4;        // 16 float4 per c-row
        const int q = i & 15;
        const float4 v = *(const float4*)(x + (size_t)(bi * kC + c) * kN + n0 + q * 4);
        xv[it] = v;
        const int nl = q * 4;
        sXT[nl + 0][c] = (_Float16)v.x;
        sXT[nl + 1][c] = (_Float16)v.y;
        sXT[nl + 2][c] = (_Float16)v.z;
        sXT[nl + 3][c] = (_Float16)v.w;
    }
    __syncthreads();

    // ---- logits[g, n] via WMMA; then Gumbel hard gate ----
    const int wave = t >> 5;
    const int lane = t & 31;
    if (wave < NT / 16) {            // 4 waves: one 16-wide n-subtile each
        const int col  = lane & 15;
        const int h    = lane >> 4;
        const int nsub = wave * 16;

        v8f acc = {};
#pragma unroll
        for (int kc = 0; kc < kC / 32; ++kc) {
            const int c0 = kc * 32;
            // A: row M=col of padded W; 16-bit A layout: halves 0..7 -> K=c0+8h+j,
            // halves 8..15 -> K=c0+16+8h+(j-8)
            const v8h a0 = *(const v8h*)&sWh[col][c0 + 8 * h];
            const v8h a1 = *(const v8h*)&sWh[col][c0 + 16 + 8 * h];
            const v16h A = __builtin_shufflevector(a0, a1,
                0, 1, 2, 3, 4, 5, 6, 7, 8, 9, 10, 11, 12, 13, 14, 15);
            // B: column n=nsub+col; halves 0..7 -> K=c0+8h+j, halves 8..15 -> K=c0+16+8h+(j-8)
            const _Float16* bp = &sXT[nsub + col][c0 + 8 * h];
            const v8h b0 = *(const v8h*)(bp);
            const v8h b1 = *(const v8h*)(bp + 16);
            const v16h Bv = __builtin_shufflevector(b0, b1,
                0, 1, 2, 3, 4, 5, 6, 7, 8, 9, 10, 11, 12, 13, 14, 15);
            acc = __builtin_amdgcn_wmma_f32_16x16x32_f16(
                false, A, false, Bv, (short)0, acc, false, false);
        }

        if (h == 0) {                 // rows M=0..3 live in acc[0..3], lanes 0..15
            const int n = n0 + nsub + col;
#pragma unroll
            for (int g = 0; g < kG; ++g) {
                const float l = acc[g] + bias[g];
                const size_t ub = ((size_t)(bi * kG + g) * kN + n) * 2;
                const float gn0 = -logf(-logf(u[ub + 0]));
                const float gn1 = -logf(-logf(u[ub + 1]));
                const float a0v =  l + gn0;      // TAU = 1
                const float a1v = -l + gn1;
                const float m  = fmaxf(a0v, a1v);
                const float e0 = expf(a0v - m);
                const float e1 = expf(a1v - m);
                const float s0 = e0 / (e0 + e1);
                const float hard = (a0v >= a1v) ? 1.0f : 0.0f;  // argmax: first on tie
                sGate[g][nsub + col] = hard - s0 + s0;           // straight-through fwd
            }
        }
    }
    __syncthreads();

    // ---- broadcast multiply + stream out: out[b][g][c][n] = x[b][c][n]*gate[b][g][n]
#pragma unroll
    for (int it = 0; it < NV4; ++it) {
        const int i = t + it * NTHREADS;
        const int c = i >> 4;
        const int q = i & 15;
        const int nl = q * 4;
        const float4 v = xv[it];
#pragma unroll
        for (int g = 0; g < kG; ++g) {
            float4 o;
            o.x = v.x * sGate[g][nl + 0];
            o.y = v.y * sGate[g][nl + 1];
            o.z = v.z * sGate[g][nl + 2];
            o.w = v.w * sGate[g][nl + 3];
            *(float4*)(out + ((size_t)((bi * kG + g) * kC + c)) * kN + n0 + nl) = o;
        }
    }
}

extern "C" void kernel_launch(void* const* d_in, const int* in_sizes, int n_in,
                              void* d_out, int out_size, void* d_ws, size_t ws_size,
                              hipStream_t stream) {
    (void)in_sizes; (void)n_in; (void)out_size; (void)d_ws; (void)ws_size;
    const float* x = (const float*)d_in[0];
    const float* W = (const float*)d_in[1];
    const float* b = (const float*)d_in[2];
    const float* u = (const float*)d_in[3];
    float* out     = (float*)d_out;

    dim3 grid(kN / NT, kB);
    dynamic_sampling_fused<<<grid, NTHREADS, 0, stream>>>(x, W, b, u, out);
}
